// Convolution_68848325755172
// MI455X (gfx1250) — matrix operand order repaired and checked by reading, not compiled, not run on hardware
//
#include <hip/hip_runtime.h>
#include <hip/hip_bf16.h>
#include <math.h>

#define N_NODES 10000
#define N_EDGES 200000

typedef __attribute__((ext_vector_type(16))) _Float16 v16h;
typedef __attribute__((ext_vector_type(8)))  _Float16 v8h;
typedef __attribute__((ext_vector_type(8)))  float    v8f;
typedef __attribute__((ext_vector_type(4)))  float    v4f;

__device__ __forceinline__ float gelu_tanh(float x) {
  const float c = 0.7978845608028654f; // sqrt(2/pi)
  float x3 = x * x * x;
  float t = tanhf(c * (x + 0.044715f * x3));
  return 0.5f * x * (1.0f + t);
}

__device__ __forceinline__ v8f wmma_f16(v16h a, v16h b, v8f c) {
  return __builtin_amdgcn_wmma_f32_16x16x32_f16(false, a, false, b, (short)0, c, false, false);
}

__device__ __forceinline__ v16h cat8(v8h a, v8h b) {
  return __builtin_shufflevector(a, b, 0, 1, 2, 3, 4, 5, 6, 7,
                                        8, 9, 10, 11, 12, 13, 14, 15);
}

// ---------------------------------------------------------------------------
// Kernel Z: zero the aggregation buffer
// ---------------------------------------------------------------------------
__global__ void zero_kernel(float* __restrict__ p, int n) {
  int i = blockIdx.x * blockDim.x + threadIdx.x;
  if (i < n) p[i] = 0.0f;
}

// ---------------------------------------------------------------------------
// Kernel 1: node linears  f0/s0 = x0 @ W * inv ; f1/s1 over the 3 vector
// components. One wave per 16-row tile (N = 625 * 16 exactly).
// ---------------------------------------------------------------------------
__global__ __launch_bounds__(32) void node_linear_kernel(
    const float* __restrict__ x,     // node_input (N,128)
    const float* __restrict__ Wf,    // W_lin0_feat (32,32)
    const float* __restrict__ Wv,    // W_lin1_feat (32,32)
    const float* __restrict__ Wsf,   // W_lin0_self (32,32)
    const float* __restrict__ Wsv,   // W_lin1_self (32,32)
    float* __restrict__ f0, float* __restrict__ f1,
    float* __restrict__ s0, float* __restrict__ s1)
{
  const int lane = threadIdx.x;
  const int row0 = blockIdx.x * 16;
  const int m    = lane & 15;
  const int akb  = (lane < 16) ? 0 : 8;
  const int bn   = lane & 15;
  const int bkb  = (lane < 16) ? 0 : 8;
  const int drow = row0 + ((lane < 16) ? 0 : 8);
  const float inv = 0.17677669529663687f;  // 1/sqrt(32)

  auto loadB32 = [&](const float* W, int t) -> v16h {
    v16h b;
    #pragma unroll
    for (int h = 0; h < 16; ++h)
      b[h] = (_Float16)W[(bkb + h + ((h >= 8) ? 8 : 0)) * 32 + t * 16 + bn];
    return b;
  };

  v8f zero = {};

  // scalar channel A fragment (16x32)
  v16h a0;
  #pragma unroll
  for (int h = 0; h < 16; ++h) {
    int k = akb + h + ((h >= 8) ? 8 : 0);
    a0[h] = (_Float16)x[(row0 + m) * 128 + k];
  }
  #pragma unroll
  for (int t = 0; t < 2; ++t) {
    v8f d = wmma_f16(a0, loadB32(Wf, t), zero);
    #pragma unroll
    for (int v = 0; v < 8; ++v) f0[(drow + v) * 32 + t * 16 + bn] = d[v] * inv;
    d = wmma_f16(a0, loadB32(Wsf, t), zero);
    #pragma unroll
    for (int v = 0; v < 8; ++v) s0[(drow + v) * 32 + t * 16 + bn] = d[v] * inv;
  }

  // vector channels, one component at a time
  for (int i = 0; i < 3; ++i) {
    v16h a1;
    #pragma unroll
    for (int h = 0; h < 16; ++h) {
      int k = akb + h + ((h >= 8) ? 8 : 0);
      a1[h] = (_Float16)x[(row0 + m) * 128 + 32 + 3 * k + i];
    }
    #pragma unroll
    for (int t = 0; t < 2; ++t) {
      v8f d = wmma_f16(a1, loadB32(Wv, t), zero);
      #pragma unroll
      for (int v = 0; v < 8; ++v)
        f1[(drow + v) * 96 + 3 * (t * 16 + bn) + i] = d[v] * inv;
      d = wmma_f16(a1, loadB32(Wsv, t), zero);
      #pragma unroll
      for (int v = 0; v < 8; ++v)
        s1[(drow + v) * 96 + 3 * (t * 16 + bn) + i] = d[v] * inv;
    }
  }
}

// ---------------------------------------------------------------------------
// Kernel 2: fused edge MLP (all three GEMMs on WMMA) + tensor-product
// messages + scatter-add. One 16-edge tile per wave per iteration, 4 waves
// per block, 5 iterations per block (12500 tiles = 625 blocks * 4 waves * 5).
// All weight B-fragments are loop-invariant and hoisted into registers.
// ---------------------------------------------------------------------------
__global__ __launch_bounds__(128) void edge_kernel(
    const float* __restrict__ edge_attr,    // (E,4)
    const float* __restrict__ edge_scalar,  // (E,8)
    const float* __restrict__ Wm1,          // (8,64)
    const float* __restrict__ Wm2,          // (64,64)
    const float* __restrict__ Wtp,          // (4,64,32)
    const float* __restrict__ f0,           // (N,32)
    const float* __restrict__ f1,           // (N,96)
    const int*   __restrict__ esrc,
    const int*   __restrict__ edst,
    float* __restrict__ agg)                // (N,256)
{
  __shared__ __align__(16) _Float16 shh1[4][16 * 64];  // h1 staging (f16)
  __shared__ __align__(16) _Float16 shh2[4][16 * 64];  // h2 staging (f16)
  __shared__ __align__(16) float    shw[4][16 * 128];  // w staging (f32)

  const int wave = threadIdx.x >> 5;
  const int lane = threadIdx.x & 31;
  _Float16* h1s = shh1[wave];
  _Float16* h2s = shh2[wave];
  float*    wsh = shw[wave];

  const int m   = lane & 15;
  const int akb = (lane < 16) ? 0 : 8;
  const int bn  = lane & 15;
  const int bkb = (lane < 16) ? 0 : 8;
  const int dr  = (lane < 16) ? 0 : 8;

  const float invr = 0.35355339059327373f;  // 1/sqrt(8)
  const float invh = 0.125f;                // 1/sqrt(64)
  v8f zero = {};

  // ---- hoisted loop-invariant B fragments ---------------------------------
  // layer 1: Wm1 (8,64) zero-padded to K=32
  v16h Bm1[4];
  #pragma unroll
  for (int t = 0; t < 4; ++t) {
    v16h b = {};
    #pragma unroll
    for (int h = 0; h < 16; ++h) {
      int k = bkb + h + ((h >= 8) ? 8 : 0);
      if (k < 8) b[h] = (_Float16)Wm1[k * 64 + t * 16 + bn];
    }
    Bm1[t] = b;
  }
  // layer 2: Wm2 (64,64), 4 col tiles x 2 K-halves
  v16h Bm2[4][2];
  #pragma unroll
  for (int t = 0; t < 4; ++t)
    #pragma unroll
    for (int hh = 0; hh < 2; ++hh) {
      v16h b;
      #pragma unroll
      for (int h = 0; h < 16; ++h)
        b[h] = (_Float16)Wm2[(hh * 32 + bkb + h + ((h >= 8) ? 8 : 0)) * 64 + t * 16 + bn];
      Bm2[t][hh] = b;
    }
  // layer 3: Wtp' (64,128) with Wtp'[x, p*32+u] = Wtp[p][x][u]
  v16h Btp[8][2];
  #pragma unroll
  for (int t = 0; t < 8; ++t)
    #pragma unroll
    for (int hh = 0; hh < 2; ++hh) {
      v16h b;
      int c = t * 16 + bn;
      int p = c >> 5, u = c & 31;
      #pragma unroll
      for (int h = 0; h < 16; ++h)
        b[h] = (_Float16)Wtp[p * 2048 + (hh * 32 + bkb + h + ((h >= 8) ? 8 : 0)) * 32 + u];
      Btp[t][hh] = b;
    }

  // ---- tile loop (do NOT unroll: barrier-separated, unrolling only bloats
  // I$ and scalar address code) --------------------------------------------
  #pragma unroll 1
  for (int it = 0; it < 5; ++it) {
    const int tile  = (it * 625 + (int)blockIdx.x) * 4 + wave;
    const int ebase = tile * 16;

    // Phase 1: h1 = gelu((s @ Wm1) * invr) via zero-padded K=32 WMMA
    {
      v4f sa = *(const v4f*)(edge_scalar + (ebase + m) * 8);
      v4f sb = *(const v4f*)(edge_scalar + (ebase + m) * 8 + 4);
      v16h a = {};
      if (lane < 16) {
        a[0] = (_Float16)sa[0]; a[1] = (_Float16)sa[1];
        a[2] = (_Float16)sa[2]; a[3] = (_Float16)sa[3];
        a[4] = (_Float16)sb[0]; a[5] = (_Float16)sb[1];
        a[6] = (_Float16)sb[2]; a[7] = (_Float16)sb[3];
      }
      #pragma unroll
      for (int t = 0; t < 4; ++t) {
        v8f d = wmma_f16(a, Bm1[t], zero);
        #pragma unroll
        for (int v = 0; v < 8; ++v)
          h1s[(dr + v) * 64 + t * 16 + bn] = (_Float16)gelu_tanh(d[v] * invr);
      }
    }
    __syncthreads();

    // Phase 2: h2 = gelu((h1 @ Wm2) * invh)
    {
      const _Float16* ap = h1s + m * 64 + akb;
      v16h alo = cat8(*(const v8h*)(ap),      *(const v8h*)(ap + 16));
      v16h ahi = cat8(*(const v8h*)(ap + 32), *(const v8h*)(ap + 48));
      #pragma unroll
      for (int t = 0; t < 4; ++t) {
        v8f d = wmma_f16(ahi, Bm2[t][1], wmma_f16(alo, Bm2[t][0], zero));
        #pragma unroll
        for (int v = 0; v < 8; ++v)
          h2s[(dr + v) * 64 + t * 16 + bn] = (_Float16)gelu_tanh(d[v] * invh);
      }
    }
    __syncthreads();

    // Phase 3: w = (h2 @ Wtp') * invh, 16x128 (f32 for message math)
    {
      const _Float16* ap = h2s + m * 64 + akb;
      v16h alo = cat8(*(const v8h*)(ap),      *(const v8h*)(ap + 16));
      v16h ahi = cat8(*(const v8h*)(ap + 32), *(const v8h*)(ap + 48));
      #pragma unroll
      for (int t = 0; t < 8; ++t) {
        v8f d = wmma_f16(ahi, Btp[t][1], wmma_f16(alo, Btp[t][0], zero));
        #pragma unroll
        for (int v = 0; v < 8; ++v)
          wsh[(dr + v) * 128 + t * 16 + bn] = d[v] * invh;
      }
    }
    __syncthreads();

    // Phase 4: messages + scatter. lane owns agg cols [lane*8, lane*8+8).
    const float INV_SQRT3 = 0.5773502691896258f;
    const int cbase = lane * 8;
    #pragma unroll 2
    for (int ee = 0; ee < 16; ++ee) {
      const int e   = ebase + ee;
      const int src = esrc[e];
      const int dst = edst[e];
      if (lane == 0 && ee + 1 < 16) {  // prefetch next gather rows
        int nsrc = esrc[e + 1];
        __builtin_prefetch(f0 + nsrc * 32, 0, 3);
        __builtin_prefetch(f1 + nsrc * 96, 0, 3);
      }
      const float a0  = edge_attr[e * 4 + 0];
      const float a1x = edge_attr[e * 4 + 1];
      const float a1y = edge_attr[e * 4 + 2];
      const float a1z = edge_attr[e * 4 + 3];
      const float* wrow = wsh + ee * 128;
      const float* e0p  = f0 + src * 32;
      const float* e1p  = f1 + src * 96;
      float* aggd = agg + dst * 256;

      if (cbase < 32) {               // m0 = w[0,u]*e0[u]*a0
        #pragma unroll
        for (int j = 0; j < 8; ++j) {
          int u = cbase + j;
          atomicAdd(&aggd[u], wrow[u] * e0p[u] * a0);
        }
      } else if (cbase < 64) {        // m1 = w[3,u]*dot(e1[u],a1)/sqrt3
        #pragma unroll
        for (int j = 0; j < 8; ++j) {
          int u = cbase + j - 32;
          float dv = e1p[3 * u] * a1x + e1p[3 * u + 1] * a1y + e1p[3 * u + 2] * a1z;
          atomicAdd(&aggd[32 + u], wrow[96 + u] * dv * INV_SQRT3);
        }
      } else if (cbase < 160) {       // m2 = w[1,u]*e0[u]*a1[i]
        #pragma unroll
        for (int j = 0; j < 8; ++j) {
          int q = cbase + j - 64;
          int u = q / 3, i = q % 3;
          float ai = (i == 0) ? a1x : ((i == 1) ? a1y : a1z);
          atomicAdd(&aggd[64 + q], wrow[32 + u] * e0p[u] * ai);
        }
      } else {                        // m3 = w[2,u]*e1[u,i]*a0
        #pragma unroll
        for (int j = 0; j < 8; ++j) {
          int q = cbase + j - 160;
          int u = q / 3, i = q % 3;
          atomicAdd(&aggd[160 + q], wrow[64 + u] * e1p[3 * u + i] * a0);
        }
      }
    }
    __syncthreads();  // wsh/h1s reused next iteration
  }
}

// ---------------------------------------------------------------------------
// Kernel 3: final node linears on aggregated messages + self-mix.
// ---------------------------------------------------------------------------
__global__ __launch_bounds__(32) void node_out_kernel(
    const float* __restrict__ agg,   // (N,256)
    const float* __restrict__ Wf0,   // (64,32)
    const float* __restrict__ Wf1,   // (64,32)
    const float* __restrict__ s0,    // (N,32)
    const float* __restrict__ s1,    // (N,96)
    float* __restrict__ out)         // (N,128)
{
  const int lane = threadIdx.x;
  const int row0 = blockIdx.x * 16;
  const int m    = lane & 15;
  const int akb  = (lane < 16) ? 0 : 8;
  const int bn   = lane & 15;
  const int bkb  = (lane < 16) ? 0 : 8;
  const int drow = row0 + ((lane < 16) ? 0 : 8);
  const float invnb = 0.22360679774997896f;  // 1/sqrt(20)
  const float COS_MIX = 0.9238795325112867f;
  const float SIN_MIX = 0.3826834323650898f;
  const float fac = SIN_MIX * 0.125f;        // * 1/sqrt(64)
  v8f zero = {};

  // conv0: g0 (16x64) @ Wf0 (64x32)
  {
    v16h alo, ahi;
    #pragma unroll
    for (int h = 0; h < 16; ++h) {
      int k = akb + h + ((h >= 8) ? 8 : 0);
      alo[h] = (_Float16)(agg[(row0 + m) * 256 + k] * invnb);
      ahi[h] = (_Float16)(agg[(row0 + m) * 256 + 32 + k] * invnb);
    }
    #pragma unroll
    for (int t = 0; t < 2; ++t) {
      v16h blo, bhi;
      #pragma unroll
      for (int h = 0; h < 16; ++h) {
        blo[h] = (_Float16)Wf0[(bkb + h + ((h >= 8) ? 8 : 0)) * 32 + t * 16 + bn];
        bhi[h] = (_Float16)Wf0[(32 + bkb + h + ((h >= 8) ? 8 : 0)) * 32 + t * 16 + bn];
      }
      v8f d = wmma_f16(ahi, bhi, wmma_f16(alo, blo, zero));
      #pragma unroll
      for (int v = 0; v < 8; ++v) {
        int col = t * 16 + bn;
        out[(drow + v) * 128 + col] =
            COS_MIX * s0[(drow + v) * 32 + col] + fac * d[v];
      }
    }
  }

  // conv1: per component i, g1[:,:,i] (16x64) @ Wf1 (64x32)
  for (int i = 0; i < 3; ++i) {
    v16h alo, ahi;
    #pragma unroll
    for (int h = 0; h < 16; ++h) {
      int k = akb + h + ((h >= 8) ? 8 : 0);
      alo[h] = (_Float16)(agg[(row0 + m) * 256 + 64 + 3 * k + i] * invnb);
      ahi[h] = (_Float16)(agg[(row0 + m) * 256 + 64 + 3 * (32 + k) + i] * invnb);
    }
    #pragma unroll
    for (int t = 0; t < 2; ++t) {
      v16h blo, bhi;
      #pragma unroll
      for (int h = 0; h < 16; ++h) {
        blo[h] = (_Float16)Wf1[(bkb + h + ((h >= 8) ? 8 : 0)) * 32 + t * 16 + bn];
        bhi[h] = (_Float16)Wf1[(32 + bkb + h + ((h >= 8) ? 8 : 0)) * 32 + t * 16 + bn];
      }
      v8f d = wmma_f16(ahi, bhi, wmma_f16(alo, blo, zero));
      #pragma unroll
      for (int v = 0; v < 8; ++v) {
        int u = t * 16 + bn;
        out[(drow + v) * 128 + 32 + 3 * u + i] =
            COS_MIX * s1[(drow + v) * 96 + 3 * u + i] + fac * d[v];
      }
    }
  }
}

// ---------------------------------------------------------------------------
extern "C" void kernel_launch(void* const* d_in, const int* in_sizes, int n_in,
                              void* d_out, int out_size, void* d_ws, size_t ws_size,
                              hipStream_t stream) {
  const float* node_input  = (const float*)d_in[0];
  const int*   edge_src    = (const int*)  d_in[1];
  const int*   edge_dst    = (const int*)  d_in[2];
  const float* edge_attr   = (const float*)d_in[3];
  const float* edge_scalar = (const float*)d_in[4];
  const float* W_lin0_feat = (const float*)d_in[5];
  const float* W_lin1_feat = (const float*)d_in[6];
  const float* W_lin0_self = (const float*)d_in[7];
  const float* W_lin1_self = (const float*)d_in[8];
  const float* W_mlp1      = (const float*)d_in[9];
  const float* W_mlp2      = (const float*)d_in[10];
  const float* W_tp        = (const float*)d_in[11];
  const float* Wf0         = (const float*)d_in[12];
  const float* Wf1         = (const float*)d_in[13];
  float* out = (float*)d_out;

  float* ws  = (float*)d_ws;
  float* f0  = ws;                   // N*32
  float* f1  = ws + N_NODES * 32;    // N*96
  float* s0  = ws + N_NODES * 128;   // N*32
  float* s1  = ws + N_NODES * 160;   // N*96
  float* agg = ws + N_NODES * 256;   // N*256

  zero_kernel<<<(N_NODES * 256 + 255) / 256, 256, 0, stream>>>(agg, N_NODES * 256);
  node_linear_kernel<<<N_NODES / 16, 32, 0, stream>>>(
      node_input, W_lin0_feat, W_lin1_feat, W_lin0_self, W_lin1_self,
      f0, f1, s0, s1);
  edge_kernel<<<625, 128, 0, stream>>>(
      edge_attr, edge_scalar, W_mlp1, W_mlp2, W_tp, f0, f1,
      edge_src, edge_dst, agg);
  node_out_kernel<<<N_NODES / 16, 32, 0, stream>>>(
      agg, Wf0, Wf1, s0, s1, out);
}
